// ImprovedCobrablock_31078383354561
// MI455X (gfx1250) — matrix operand order repaired and checked by reading, not compiled, not run on hardware
//
#include <hip/hip_runtime.h>

// ---------------- problem constants (from reference) ----------------
#define BATCH  4
#define SEQ    2048
#define DMODEL 384
#define INNER  768
#define NSTATE 16
#define KCONV  4
#define FFN    1024
#define NTOK   (BATCH * SEQ)            // 8192 tokens
#define CHUNKS 16
#define CLEN   (SEQ / CHUNKS)           // 128

typedef unsigned short u16;
typedef __attribute__((ext_vector_type(16))) __bf16 v16bf;
typedef __attribute__((ext_vector_type(8)))  float  v8f;
typedef __attribute__((ext_vector_type(16))) u16    v16u;
typedef __attribute__((ext_vector_type(8)))  u16    v8u;

__device__ __forceinline__ u16 f2bf(float f) {  // RNE f32 -> bf16
  unsigned u = __builtin_bit_cast(unsigned, f);
  u += 0x7FFFu + ((u >> 16) & 1u);
  return (u16)(u >> 16);
}
__device__ __forceinline__ float silu_f(float v) {
  return v / (1.0f + __expf(-v));
}

// ---------------- weight convert + transpose: fp32[K,N] -> bf16[N,K] ----
__global__ __launch_bounds__(256) void wt_convert_t(
    const float* __restrict__ W, u16* __restrict__ Wt, int K, int N) {
  size_t idx = (size_t)blockIdx.x * 256 + threadIdx.x;
  if (idx >= (size_t)K * N) return;
  int n = (int)(idx % N);
  int k = (int)(idx / N);
  Wt[(size_t)n * K + k] = f2bf(W[idx]);
}

// ---------------- WMMA fragment helpers (CDNA5 wave32 layouts) ------------
// A 16x32 bf16: lane<16 holds M=lane, K = base+0..15; lanes>=16 K = base+16..31
__device__ __forceinline__ v16bf load_a_frag(const u16* __restrict__ row,
                                             int k, int half) {
  v8u a0 = *(const v8u*)(row + k + 16 * half);
  v8u a1 = *(const v8u*)(row + k + 16 * half + 8);
  v16u av = __builtin_shufflevector(a0, a1,
      0, 1, 2, 3, 4, 5, 6, 7, 8, 9, 10, 11, 12, 13, 14, 15);
  return __builtin_bit_cast(v16bf, av);
}
// B 32x16 bf16 (from [N,K] transposed weights): lane&15 = N col, 16 contiguous K
__device__ __forceinline__ v16bf load_b_frag(const u16* __restrict__ row,
                                             int k, int half) {
  return __builtin_bit_cast(v16bf, *(const v16u*)(row + k + 16 * half));
}
// C/D f32 16x16: lane -> column, VGPR r -> row r + 8*half
__device__ __forceinline__ void store_c_tile(float* __restrict__ C, int N,
                                             int m0, int n, int half,
                                             const v8f& acc) {
  float* p = C + (size_t)(m0 + 8 * half) * N + n;
  #pragma unroll
  for (int r = 0; r < 8; ++r) p[(size_t)r * N] = acc[r];
}

// ---------------- big WMMA GEMM: wave tile 32x64, block 64x128 ------------
// 128 threads = 4 waves (2M x 2N).  Per 32-K step each wave: 2 A-frags,
// 4 B-frags, 8 WMMAs -> 0.75KB of fragment loads per WMMA (vs 2KB naive).
// Call sites guarantee K%32==0, N%128==0, M%64==0 -> no guards, EXEC all-1s.
__global__ __launch_bounds__(128) void gemm_bf16_wmma_big(
    const u16* __restrict__ A, const u16* __restrict__ Bt,
    float* __restrict__ C, int M, int N, int K) {
  const int lane  = threadIdx.x & 31;
  const int wave  = threadIdx.x >> 5;
  const int mBase = blockIdx.y * 64 + (wave >> 1) * 32;
  const int nBase = blockIdx.x * 128 + (wave & 1) * 64;
  const int half = lane >> 4;
  const int lid  = lane & 15;
  const u16* arow0 = A + (size_t)(mBase + lid) * K;
  const u16* arow1 = arow0 + (size_t)16 * K;
  const u16* brow[4];
  #pragma unroll
  for (int j = 0; j < 4; ++j)
    brow[j] = Bt + (size_t)(nBase + 16 * j + lid) * K;
  v8f acc0[4], acc1[4];
  #pragma unroll
  for (int j = 0; j < 4; ++j) {
    acc0[j] = (v8f){0.f, 0.f, 0.f, 0.f, 0.f, 0.f, 0.f, 0.f};
    acc1[j] = acc0[j];
  }
  for (int k = 0; k < K; k += 32) {
    // A streams from HBM (activations / im2col): WGP-scope prefetch one
    // K-step ahead (locality 3 -> scope 0: pull into all cache levels).
    __builtin_prefetch(arow0 + k + 32, 0, 3);
    __builtin_prefetch(arow1 + k + 32, 0, 3);
    v16bf a0 = load_a_frag(arow0, k, half);
    v16bf a1 = load_a_frag(arow1, k, half);
    #pragma unroll
    for (int j = 0; j < 4; ++j) {
      v16bf b = load_b_frag(brow[j], k, half);
      acc0[j] = __builtin_amdgcn_wmma_f32_16x16x32_bf16(
          false, a0, false, b, (short)0, acc0[j], false, false);
      acc1[j] = __builtin_amdgcn_wmma_f32_16x16x32_bf16(
          false, a1, false, b, (short)0, acc1[j], false, false);
    }
  }
  #pragma unroll
  for (int j = 0; j < 4; ++j) {
    store_c_tile(C, N, mBase,      nBase + 16 * j + lid, half, acc0[j]);
    store_c_tile(C, N, mBase + 16, nBase + 16 * j + lid, half, acc1[j]);
  }
}

// ---------------- small-N WMMA GEMM (x_proj, N=32): wave tile 32x32 -------
__global__ __launch_bounds__(128) void gemm_bf16_wmma_small(
    const u16* __restrict__ A, const u16* __restrict__ Bt,
    float* __restrict__ C, int M, int N, int K) {
  const int lane  = threadIdx.x & 31;
  const int wave  = threadIdx.x >> 5;
  const int mBase = blockIdx.y * 64 + (wave >> 1) * 32;
  const int nBase = blockIdx.x * 64 + (wave & 1) * 32;
  if (mBase >= M || nBase >= N) return;            // wave-uniform guard
  const int half = lane >> 4;
  const int lid  = lane & 15;
  const u16* arow0 = A  + (size_t)(mBase + lid) * K;
  const u16* arow1 = arow0 + (size_t)16 * K;
  const u16* brow0 = Bt + (size_t)(nBase + lid) * K;
  const u16* brow1 = brow0 + (size_t)16 * K;
  v8f acc00 = {0.f, 0.f, 0.f, 0.f, 0.f, 0.f, 0.f, 0.f};
  v8f acc01 = acc00, acc10 = acc00, acc11 = acc00;
  for (int k = 0; k < K; k += 32) {
    v16bf a0 = load_a_frag(arow0, k, half);
    v16bf a1 = load_a_frag(arow1, k, half);
    v16bf b0 = load_b_frag(brow0, k, half);
    v16bf b1 = load_b_frag(brow1, k, half);
    acc00 = __builtin_amdgcn_wmma_f32_16x16x32_bf16(false, a0, false, b0,
                                                    (short)0, acc00, false, false);
    acc01 = __builtin_amdgcn_wmma_f32_16x16x32_bf16(false, a0, false, b1,
                                                    (short)0, acc01, false, false);
    acc10 = __builtin_amdgcn_wmma_f32_16x16x32_bf16(false, a1, false, b0,
                                                    (short)0, acc10, false, false);
    acc11 = __builtin_amdgcn_wmma_f32_16x16x32_bf16(false, a1, false, b1,
                                                    (short)0, acc11, false, false);
  }
  store_c_tile(C, N, mBase,      nBase      + lid, half, acc00);
  store_c_tile(C, N, mBase,      nBase + 16 + lid, half, acc01);
  store_c_tile(C, N, mBase + 16, nBase      + lid, half, acc10);
  store_c_tile(C, N, mBase + 16, nBase + 16 + lid, half, acc11);
}

// ---------------- RMSNorm (wave per row) -> bf16 row ----------------
__global__ __launch_bounds__(256) void rmsnorm_bf16(
    const float* __restrict__ x, const float* __restrict__ w,
    u16* __restrict__ out, int D) {
  int row  = blockIdx.x * 8 + (threadIdx.x >> 5);
  int lane = threadIdx.x & 31;
  const float* xr = x + (size_t)row * D;
  float s = 0.f;
  for (int i = lane; i < D; i += 32) { float v = xr[i]; s += v * v; }
  #pragma unroll
  for (int off = 16; off; off >>= 1) s += __shfl_xor(s, off, 32);
  float inv = rsqrtf(s / (float)D + 1e-6f);
  u16* orow = out + (size_t)row * D;
  for (int i = lane; i < D; i += 32) orow[i] = f2bf(xr[i] * inv * w[i]);
}

// ---------------- LayerNorm(out)*gamma+beta + residual ----------------
__global__ __launch_bounds__(256) void ln_residual(
    const float* __restrict__ o, const float* __restrict__ resid,
    const float* __restrict__ gamma, const float* __restrict__ beta,
    float* __restrict__ x2, int D) {
  int row  = blockIdx.x * 8 + (threadIdx.x >> 5);
  int lane = threadIdx.x & 31;
  const float* orow = o + (size_t)row * D;
  float s1 = 0.f, s2 = 0.f;
  for (int i = lane; i < D; i += 32) { float v = orow[i]; s1 += v; s2 += v * v; }
  #pragma unroll
  for (int off = 16; off; off >>= 1) {
    s1 += __shfl_xor(s1, off, 32);
    s2 += __shfl_xor(s2, off, 32);
  }
  float mu   = s1 / (float)D;
  float var  = s2 / (float)D - mu * mu;
  float rinv = rsqrtf(var + 1e-5f);
  const float* rrow = resid + (size_t)row * D;
  float* xrow = x2 + (size_t)row * D;
  for (int i = lane; i < D; i += 32)
    xrow[i] = rrow[i] + (orow[i] - mu) * rinv * gamma[i] + beta[i];
}

// ---------------- im2col for conv1d (k=4, pad left 1 / right 2) ----------
__global__ __launch_bounds__(256) void im2col_conv(
    const float* __restrict__ xz, u16* __restrict__ col) {
  size_t idx = (size_t)blockIdx.x * 256 + threadIdx.x;
  const size_t total = (size_t)NTOK * (KCONV * INNER);
  if (idx >= total) return;
  int    c = (int)(idx % (KCONV * INNER));
  size_t r = idx / (KCONV * INNER);
  int w = c / INNER, i = c - w * INNER;
  int t = (int)(r % SEQ);
  int b = (int)(r / SEQ);
  int ts = t - 1 + w;                       // SAME pad for k=4 -> (1,2)
  float v = 0.f;
  if (ts >= 0 && ts < SEQ)
    v = xz[((size_t)b * SEQ + ts) * (2 * INNER) + i];
  col[idx] = f2bf(v);
}

// ---------------- conv bias + SiLU; write fp32 + bf16 copies -------------
__global__ __launch_bounds__(256) void conv_bias_silu(
    float* __restrict__ xc, const float* __restrict__ bias,
    u16* __restrict__ xbf) {
  size_t idx = (size_t)blockIdx.x * 256 + threadIdx.x;
  if (idx >= (size_t)NTOK * INNER) return;
  int c = (int)(idx % INNER);
  float s = silu_f(xc[idx] + bias[c]);
  xc[idx]  = s;
  xbf[idx] = f2bf(s);
}

// ---------------- delta -> softplus(delta + dt_bias) in-place -------------
__global__ __launch_bounds__(256) void sp_prep(
    float* __restrict__ delta, const float* __restrict__ dtb) {
  size_t idx = (size_t)blockIdx.x * 256 + threadIdx.x;
  if (idx >= (size_t)NTOK * INNER) return;
  int d = (int)(idx % INNER);
  float v = delta[idx] + dtb[d];
  delta[idx] = (v > 20.f) ? v : __logf(1.f + __expf(v));
}

// ================= chunked selective scan (3 phases) ======================
// Diagonal SSM: per (b,d,n), h_t = a_t*h_{t-1} + u_t is a scalar linear
// recurrence.  Chunked: serial depth 2048 -> 128+16+128, threads 3072 -> 49152.

// phase 1: thread = (b, chunk, d) -> chunk-local state + transition product
__global__ __launch_bounds__(256) void scan_phase1(
    const float* __restrict__ sp, const float* __restrict__ xconv,
    const float* __restrict__ bc, const float* __restrict__ A_log,
    float* __restrict__ hc, float* __restrict__ pc) {
  int idx = blockIdx.x * 256 + threadIdx.x;
  if (idx >= BATCH * CHUNKS * INNER) return;
  int d = idx % INNER;
  int c = (idx / INNER) % CHUNKS;
  int b = idx / (INNER * CHUNKS);
  float a[NSTATE], h[NSTATE], p[NSTATE];
  #pragma unroll
  for (int n = 0; n < NSTATE; ++n) {
    a[n] = -__expf(A_log[d * NSTATE + n]);
    h[n] = 0.f;
    p[n] = 1.f;
  }
  for (int t = 0; t < CLEN; ++t) {
    size_t row = (size_t)b * SEQ + c * CLEN + t;
    float spv = sp[row * INNER + d];
    float xv  = xconv[row * INNER + d];
    const float* bcr = bc + row * (2 * NSTATE);
    #pragma unroll
    for (int n = 0; n < NSTATE; ++n) {
      float at = __expf(spv * a[n]);
      h[n] = at * h[n] + (spv * bcr[n]) * xv;
      p[n] *= at;
    }
  }
  size_t o = ((size_t)(b * CHUNKS + c) * INNER + d) * NSTATE;
  #pragma unroll
  for (int n = 0; n < NSTATE; ++n) { hc[o + n] = h[n]; pc[o + n] = p[n]; }
}

// phase 2: thread = (b, d, n); 16-step sequential compose across chunks
__global__ __launch_bounds__(256) void scan_phase2(
    const float* __restrict__ hc, const float* __restrict__ pc,
    float* __restrict__ hin) {
  int idx = blockIdx.x * 256 + threadIdx.x;
  if (idx >= BATCH * INNER * NSTATE) return;
  int n = idx % NSTATE;
  int d = (idx / NSTATE) % INNER;
  int b = idx / (NSTATE * INNER);
  float s = 0.f;
  for (int c = 0; c < CHUNKS; ++c) {
    size_t o = ((size_t)(b * CHUNKS + c) * INNER + d) * NSTATE + n;
    hin[o] = s;
    s = pc[o] * s + hc[o];
  }
}

// phase 3: replay chunk from true incoming state; fused D*x, silu(z) gate,
// bf16 conversion for the out_proj GEMM input.
__global__ __launch_bounds__(256) void scan_phase3(
    const float* __restrict__ sp, const float* __restrict__ xconv,
    const float* __restrict__ bc, const float* __restrict__ A_log,
    const float* __restrict__ hin, const float* __restrict__ Dp,
    const float* __restrict__ xz, u16* __restrict__ ybf) {
  int idx = blockIdx.x * 256 + threadIdx.x;
  if (idx >= BATCH * CHUNKS * INNER) return;
  int d = idx % INNER;
  int c = (idx / INNER) % CHUNKS;
  int b = idx / (INNER * CHUNKS);
  float a[NSTATE], h[NSTATE];
  size_t o = ((size_t)(b * CHUNKS + c) * INNER + d) * NSTATE;
  #pragma unroll
  for (int n = 0; n < NSTATE; ++n) {
    a[n] = -__expf(A_log[d * NSTATE + n]);
    h[n] = hin[o + n];
  }
  float dval = Dp[d];
  for (int t = 0; t < CLEN; ++t) {
    size_t row = (size_t)b * SEQ + c * CLEN + t;
    float spv = sp[row * INNER + d];
    float xv  = xconv[row * INNER + d];
    const float* bcr = bc + row * (2 * NSTATE);
    float yt = 0.f;
    #pragma unroll
    for (int n = 0; n < NSTATE; ++n) {
      h[n] = __expf(spv * a[n]) * h[n] + (spv * bcr[n]) * xv;
      yt  += h[n] * bcr[NSTATE + n];
    }
    yt += dval * xv;
    float z = xz[row * (2 * INNER) + INNER + d];
    ybf[row * INNER + d] = f2bf(yt * silu_f(z));
  }
}

// ---------------- FFN: silu(gate) * up -> bf16 ----------------------------
__global__ __launch_bounds__(256) void silu_mul(
    const float* __restrict__ g, const float* __restrict__ u,
    u16* __restrict__ hid) {
  size_t idx = (size_t)blockIdx.x * 256 + threadIdx.x;
  if (idx >= (size_t)NTOK * FFN) return;
  hid[idx] = f2bf(silu_f(g[idx]) * u[idx]);
}

// ---------------- final residual add --------------------------------------
__global__ __launch_bounds__(256) void final_add(
    const float* __restrict__ x2, const float* __restrict__ dn,
    float* __restrict__ out) {
  size_t idx = (size_t)blockIdx.x * 256 + threadIdx.x;
  if (idx >= (size_t)NTOK * DMODEL) return;
  out[idx] = x2[idx] + dn[idx];
}

// ==========================================================================
extern "C" void kernel_launch(void* const* d_in, const int* in_sizes, int n_in,
                              void* d_out, int out_size, void* d_ws, size_t ws_size,
                              hipStream_t stream) {
  (void)in_sizes; (void)n_in; (void)out_size; (void)ws_size;
  const float* x       = (const float*)d_in[0];
  const float* rms1_w  = (const float*)d_in[1];
  const float* rms2_w  = (const float*)d_in[2];
  const float* in_proj = (const float*)d_in[3];
  const float* conv_w  = (const float*)d_in[4];   // [4, INNER, INNER] flat
  const float* conv_b  = (const float*)d_in[5];
  const float* x_proj  = (const float*)d_in[6];
  const float* dt_w    = (const float*)d_in[7];
  const float* dt_b    = (const float*)d_in[8];
  const float* A_log   = (const float*)d_in[9];
  const float* D_par   = (const float*)d_in[10];
  const float* out_w   = (const float*)d_in[11];
  const float* gamma   = (const float*)d_in[12];
  const float* beta    = (const float*)d_in[13];
  const float* gate_w  = (const float*)d_in[14];
  const float* up_w    = (const float*)d_in[15];
  const float* down_w  = (const float*)d_in[16];
  float* out = (float*)d_out;

  // ---- workspace allocator (256B aligned slabs) ----
  char* ws = (char*)d_ws;
  size_t off = 0;
  auto alloc = [&](size_t bytes) -> void* {
    off = (off + 255) & ~(size_t)255;
    void* p = ws + off;
    off += bytes;
    return p;
  };
  u16*   hbf     = (u16*)  alloc((size_t)NTOK * DMODEL * 2);
  u16*   w_in    = (u16*)  alloc((size_t)DMODEL * 2 * INNER * 2);
  u16*   w_conv  = (u16*)  alloc((size_t)KCONV * INNER * INNER * 2);
  u16*   w_xp    = (u16*)  alloc((size_t)INNER * 2 * NSTATE * 2);
  u16*   w_dt    = (u16*)  alloc((size_t)INNER * INNER * 2);
  u16*   w_out   = (u16*)  alloc((size_t)INNER * DMODEL * 2);
  u16*   w_gate  = (u16*)  alloc((size_t)DMODEL * FFN * 2);
  u16*   w_up    = (u16*)  alloc((size_t)DMODEL * FFN * 2);
  u16*   w_down  = (u16*)  alloc((size_t)FFN * DMODEL * 2);
  float* xz      = (float*)alloc((size_t)NTOK * 2 * INNER * 4);
  u16*   colbuf  = (u16*)  alloc((size_t)NTOK * KCONV * INNER * 2);
  float* xconv   = (float*)alloc((size_t)NTOK * INNER * 4);
  u16*   xconvbf = (u16*)  alloc((size_t)NTOK * INNER * 2);
  float* bc      = (float*)alloc((size_t)NTOK * 2 * NSTATE * 4);
  float* delta   = (float*)alloc((size_t)NTOK * INNER * 4);   // becomes softplus(delta)
  float* hcbuf   = (float*)alloc((size_t)BATCH * CHUNKS * INNER * NSTATE * 4);
  float* pcbuf   = (float*)alloc((size_t)BATCH * CHUNKS * INNER * NSTATE * 4);
  float* hinbuf  = (float*)alloc((size_t)BATCH * CHUNKS * INNER * NSTATE * 4);
  u16*   ybf     = (u16*)  alloc((size_t)NTOK * INNER * 2);
  float* mout    = (float*)alloc((size_t)NTOK * DMODEL * 4);
  float* x2      = (float*)alloc((size_t)NTOK * DMODEL * 4);
  u16*   x2bf    = (u16*)  alloc((size_t)NTOK * DMODEL * 2);
  float* gbuf    = (float*)alloc((size_t)NTOK * FFN * 4);
  float* ubuf    = (float*)alloc((size_t)NTOK * FFN * 4);
  u16*   hidbf   = (u16*)  alloc((size_t)NTOK * FFN * 2);
  float* dnout   = (float*)alloc((size_t)NTOK * DMODEL * 4);

  auto ew_grid = [](size_t total) { return dim3((unsigned)((total + 255) / 256)); };
  auto cvt = [&](const float* W, u16* Wt, int K, int N) {
    wt_convert_t<<<ew_grid((size_t)K * N), 256, 0, stream>>>(W, Wt, K, N);
  };
  auto gemm = [&](const u16* A, const u16* Bt, float* C, int M, int N, int K) {
    if (N % 128 == 0)
      gemm_bf16_wmma_big<<<dim3(N / 128, M / 64), 128, 0, stream>>>(
          A, Bt, C, M, N, K);
    else
      gemm_bf16_wmma_small<<<dim3((N + 63) / 64, (M + 63) / 64), 128, 0, stream>>>(
          A, Bt, C, M, N, K);
  };

  // 1. Weights -> transposed bf16
  cvt(in_proj, w_in,   DMODEL,        2 * INNER);
  cvt(conv_w,  w_conv, KCONV * INNER, INNER);   // flat [w*I, O] matches im2col K order
  cvt(x_proj,  w_xp,   INNER,         2 * NSTATE);
  cvt(dt_w,    w_dt,   INNER,         INNER);
  cvt(out_w,   w_out,  INNER,         DMODEL);
  cvt(gate_w,  w_gate, DMODEL,        FFN);
  cvt(up_w,    w_up,   DMODEL,        FFN);
  cvt(down_w,  w_down, FFN,           DMODEL);

  // 2. h = rms_norm(x, rms1_w) -> bf16
  rmsnorm_bf16<<<NTOK / 8, 256, 0, stream>>>(x, rms1_w, hbf, DMODEL);

  // 3. xz = h @ in_proj_w
  gemm(hbf, w_in, xz, NTOK, 2 * INNER, DMODEL);

  // 4. conv1d as im2col + GEMM + bias/SiLU
  im2col_conv<<<ew_grid((size_t)NTOK * KCONV * INNER), 256, 0, stream>>>(xz, colbuf);
  gemm(colbuf, w_conv, xconv, NTOK, INNER, KCONV * INNER);
  conv_bias_silu<<<ew_grid((size_t)NTOK * INNER), 256, 0, stream>>>(xconv, conv_b, xconvbf);

  // 5. B,C and delta
  gemm(xconvbf, w_xp, bc,    NTOK, 2 * NSTATE, INNER);
  gemm(xconvbf, w_dt, delta, NTOK, INNER,      INNER);
  sp_prep<<<ew_grid((size_t)NTOK * INNER), 256, 0, stream>>>(delta, dt_b);

  // 6. chunked selective scan (phase3 fuses D*x + silu(z) gate + bf16 cvt)
  const int scan_threads = BATCH * CHUNKS * INNER;       // 49152
  scan_phase1<<<scan_threads / 256, 256, 0, stream>>>(
      delta, xconv, bc, A_log, hcbuf, pcbuf);
  scan_phase2<<<(BATCH * INNER * NSTATE) / 256, 256, 0, stream>>>(
      hcbuf, pcbuf, hinbuf);
  scan_phase3<<<scan_threads / 256, 256, 0, stream>>>(
      delta, xconv, bc, A_log, hinbuf, D_par, xz, ybf);

  // 7. out_proj; x2 = x + LN(out)
  gemm(ybf, w_out, mout, NTOK, DMODEL, INNER);
  ln_residual<<<NTOK / 8, 256, 0, stream>>>(mout, x, gamma, beta, x2, DMODEL);

  // 8. FFN
  rmsnorm_bf16<<<NTOK / 8, 256, 0, stream>>>(x2, rms2_w, x2bf, DMODEL);
  gemm(x2bf, w_gate, gbuf, NTOK, FFN, DMODEL);
  gemm(x2bf, w_up,   ubuf, NTOK, FFN, DMODEL);
  silu_mul<<<ew_grid((size_t)NTOK * FFN), 256, 0, stream>>>(gbuf, ubuf, hidbf);
  gemm(hidbf, w_down, dnout, NTOK, DMODEL, FFN);
  final_add<<<ew_grid((size_t)NTOK * DMODEL), 256, 0, stream>>>(x2, dnout, out);
}
